// KernelSVM_57269093924935
// MI455X (gfx1250) — compile-verified, matrix operand order checked by reading, not run on hardware
//
#include <hip/hip_runtime.h>
#include <hip/hip_bf16.h>

typedef __attribute__((ext_vector_type(16))) __bf16 v16bf;
typedef __attribute__((ext_vector_type(8)))  float  v8f;
typedef __attribute__((ext_vector_type(4)))  int    v4i;
typedef unsigned short ushort_t;

#define DDIM     512
#define TILE_M   64          // rows per WG (4 m-supertiles x 16)
#define TILE_N   64          // cols per WG pass (2 n-supertiles x 32)
#define NCHUNKS  8
#define T_STRIDE 520         // 512 + 8 halfs pad -> 1040B row stride (4-bank rotate)

// ---- async global->LDS b128 copy (CDNA5 path), with safe fallback ----
__device__ __forceinline__ void stage_b128(const uint4* g, uint4* l) {
#if __has_builtin(__builtin_amdgcn_global_load_async_to_lds_b128)
    __builtin_amdgcn_global_load_async_to_lds_b128(
        (__attribute__((address_space(1))) v4i*)g,
        (__attribute__((address_space(3))) v4i*)l, 0, 0);
#else
    *l = *g;
#endif
}

__device__ __forceinline__ void wait_async() {
#if __has_builtin(__builtin_amdgcn_s_wait_asynccnt)
    __builtin_amdgcn_s_wait_asynccnt(0);
#else
    asm volatile("s_wait_asynccnt 0" ::: "memory");
#endif
}

// ---- f32 -> (bf16 hi, bf16 lo) split (RNE rounding via bit trick) ----
__global__ void split_bf16_kernel(const float* __restrict__ src,
                                  ushort_t* __restrict__ hi,
                                  ushort_t* __restrict__ lo, int n) {
    int i = blockIdx.x * 256 + threadIdx.x;
    if (i >= n) return;
    float f = src[i];
    unsigned u  = __float_as_uint(f);
    unsigned rh = (u + 0x7FFFu + ((u >> 16) & 1u)) >> 16;
    hi[i] = (ushort_t)rh;
    float fh = __uint_as_float(rh << 16);
    float l  = f - fh;
    unsigned u2 = __float_as_uint(l);
    unsigned rl = (u2 + 0x7FFFu + ((u2 >> 16) & 1u)) >> 16;
    lo[i] = (ushort_t)rl;
}

// ---- exact f32 row squared-norms: one wave32 per row ----
__global__ void rownorm_kernel(const float* __restrict__ src,
                               float* __restrict__ out, int rows) {
    int row  = blockIdx.x * 8 + (threadIdx.x >> 5);
    int lane = threadIdx.x & 31;
    if (row >= rows) return;
    const float4* p = (const float4*)(src + (size_t)row * DDIM);
    float s = 0.0f;
    for (int j = lane; j < DDIM / 4; j += 32) {
        float4 v = p[j];
        s += v.x * v.x + v.y * v.y + v.z * v.z + v.w * v.w;
    }
    s += __shfl_xor(s, 16); s += __shfl_xor(s, 8);
    s += __shfl_xor(s, 4);  s += __shfl_xor(s, 2); s += __shfl_xor(s, 1);
    if (lane == 0) out[row] = s;
}

// ---- out[m] = bias (atomics accumulate on top; re-run every launch) ----
__global__ void init_out_kernel(float* __restrict__ out,
                                const float* __restrict__ bias, int m) {
    int i = blockIdx.x * 256 + threadIdx.x;
    if (i < m) out[i] = bias[0];
}

// ---- LDS tile fetch matching WMMA bf16 A/B per-lane K packing ----
__device__ __forceinline__ v16bf load_tile(const ushort_t* lds, int row,
                                           int lane, int kc) {
    int kbase = kc * 32 + ((lane >> 4) << 3);
    const ushort_t* p = lds + row * T_STRIDE + kbase;
    union { v16bf v; uint4 q[2]; } u;
    u.q[0] = *(const uint4*)(p);
    u.q[1] = *(const uint4*)(p + 16);
    return u.v;
}

#define WMMA_BF16(A, B, C) \
    __builtin_amdgcn_wmma_f32_16x16x32_bf16(false, (A), false, (B), (short)0, (C), false, false)

__global__ __launch_bounds__(256)
void svm_main_kernel(const ushort_t* __restrict__ xhi, const ushort_t* __restrict__ xlo,
                     const ushort_t* __restrict__ yhi, const ushort_t* __restrict__ ylo,
                     const float* __restrict__ xx,  const float* __restrict__ yy,
                     const float* __restrict__ weight, const float* __restrict__ gamma,
                     float* __restrict__ out, int M, int N) {
    extern __shared__ ushort_t smem[];
    ushort_t* Ah = smem;                         // 64 x 520 halfs each
    ushort_t* Al = Ah + TILE_M * T_STRIDE;
    ushort_t* Bh = Al + TILE_M * T_STRIDE;
    ushort_t* Bl = Bh + TILE_N * T_STRIDE;       // total 260 KB

    const int tid  = threadIdx.x;
    const int lane = tid & 31;
    const int wave = tid >> 5;
    const int ws_m = wave & 3;                   // 4 m-supertiles (16 rows)
    const int ws_n = wave >> 2;                  // 2 n-supertiles (32 cols)

    const int nchunk_len = N / NCHUNKS;          // 1024
    const int mblk   = blockIdx.x / NCHUNKS;
    const int nchunk = blockIdx.x % NCHUNKS;
    const int mrow0  = mblk * TILE_M;
    const int n0base = nchunk * nchunk_len;

    // ---- stage A (64 rows, hi+lo) into padded LDS once (async) ----
    {
        const uint4* gAh = (const uint4*)xhi;
        const uint4* gAl = (const uint4*)xlo;
        uint4* sAh = (uint4*)Ah;
        uint4* sAl = (uint4*)Al;
        for (int i = tid; i < TILE_M * 64; i += 256) {
            int row = i >> 6, j = i & 63;        // 64 uint4 data per row
            size_t src = (size_t)(mrow0 + row) * 64 + j;
            stage_b128(gAh + src, sAh + row * 65 + j);
            stage_b128(gAl + src, sAl + row * 65 + j);
        }
        wait_async();
    }

    // C-layout: VGPR v, lanes 0-15 -> M=v; lanes 16-31 -> M=v+8
    const int mlanehalf = (lane >> 4) * 8;
    float xxv[8];
#pragma unroll
    for (int v = 0; v < 8; ++v)
        xxv[v] = xx[mrow0 + ws_m * 16 + v + mlanehalf];
    const float g = gamma[0];
    float acc[8];
#pragma unroll
    for (int v = 0; v < 8; ++v) acc[v] = 0.0f;

    const int arow  = ws_m * 16 + (lane & 15);
    const int brow1 = ws_n * 32 + (lane & 15);
    const int brow2 = brow1 + 16;

    const int passes = nchunk_len / TILE_N;      // 16
    for (int p = 0; p < passes; ++p) {
        const int n0 = n0base + p * TILE_N;

        __syncthreads();                         // prior B fully consumed
        {
            const uint4* gBh = (const uint4*)yhi;
            const uint4* gBl = (const uint4*)ylo;
            uint4* sBh = (uint4*)Bh;
            uint4* sBl = (uint4*)Bl;
            for (int i = tid; i < TILE_N * 64; i += 256) {
                int row = i >> 6, j = i & 63;    // 64 uint4 data per col row
                size_t src = (size_t)(n0 + row) * 64 + j;
                stage_b128(gBh + src, sBh + row * 65 + j);
                stage_b128(gBl + src, sBl + row * 65 + j);
            }
            wait_async();
        }
        __syncthreads();

        // ---- software-pipelined k-loop: double-buffered tile sets ----
        // t[.]: 0=b1h 1=b2h 2=al 3=ah 4=b1l 5=b2l
        v8f c0 = {}, c1 = {};
        v16bf t0[6], t1[6];
        t0[0] = load_tile(Bh, brow1, lane, 0);
        t0[1] = load_tile(Bh, brow2, lane, 0);
        t0[2] = load_tile(Al, arow,  lane, 0);
        t0[3] = load_tile(Ah, arow,  lane, 0);
        t0[4] = load_tile(Bl, brow1, lane, 0);
        t0[5] = load_tile(Bl, brow2, lane, 0);
#pragma unroll
        for (int kc = 0; kc < 16; ++kc) {
            v16bf (&cur)[6] = (kc & 1) ? t1 : t0;
            v16bf (&nxt)[6] = (kc & 1) ? t0 : t1;
            if (kc < 15) {                       // prefetch next kc's tiles
                nxt[0] = load_tile(Bh, brow1, lane, kc + 1);
                nxt[1] = load_tile(Bh, brow2, lane, kc + 1);
                nxt[2] = load_tile(Al, arow,  lane, kc + 1);
                nxt[3] = load_tile(Ah, arow,  lane, kc + 1);
                nxt[4] = load_tile(Bl, brow1, lane, kc + 1);
                nxt[5] = load_tile(Bl, brow2, lane, kc + 1);
            }
            c0 = WMMA_BF16(cur[2], cur[0], c0);  // al  x b1h
            c1 = WMMA_BF16(cur[2], cur[1], c1);  // al  x b2h
            c0 = WMMA_BF16(cur[3], cur[0], c0);  // ah  x b1h
            c1 = WMMA_BF16(cur[3], cur[1], c1);  // ah  x b2h
            c0 = WMMA_BF16(cur[3], cur[4], c0);  // ah  x b1l
            c1 = WMMA_BF16(cur[3], cur[5], c1);  // ah  x b2l
        }

        // ---- fused epilogue: exp(-g*sq)*w, accumulate per row ----
        const int ncol0 = n0 + ws_n * 32 + (lane & 15);
        {
            const float yv = yy[ncol0];
            const float wv = weight[ncol0];
#pragma unroll
            for (int v = 0; v < 8; ++v)
                acc[v] += __expf(-g * fmaxf(xxv[v] + yv - 2.0f * c0[v], 0.0f)) * wv;
        }
        {
            const float yv = yy[ncol0 + 16];
            const float wv = weight[ncol0 + 16];
#pragma unroll
            for (int v = 0; v < 8; ++v)
                acc[v] += __expf(-g * fmaxf(xxv[v] + yv - 2.0f * c1[v], 0.0f)) * wv;
        }
    }

    // ---- reduce 16 n-lanes per row half, atomically add partials ----
#pragma unroll
    for (int v = 0; v < 8; ++v) {
        float s = acc[v];
        s += __shfl_xor(s, 1); s += __shfl_xor(s, 2);
        s += __shfl_xor(s, 4); s += __shfl_xor(s, 8);
        if ((lane & 15) == 0)
            atomicAdd(&out[mrow0 + ws_m * 16 + v + mlanehalf], s);
    }
}

extern "C" void kernel_launch(void* const* d_in, const int* in_sizes, int n_in,
                              void* d_out, int out_size, void* d_ws, size_t ws_size,
                              hipStream_t stream) {
    const float* x      = (const float*)d_in[0];
    const float* xt     = (const float*)d_in[1];
    const float* gamma  = (const float*)d_in[2];
    const float* weight = (const float*)d_in[3];
    const float* bias   = (const float*)d_in[4];
    float* out = (float*)d_out;

    const int MD = in_sizes[0];
    const int ND = in_sizes[1];
    const int N  = in_sizes[3];
    const int M  = MD / DDIM;

    // workspace layout: x_hi | x_lo | y_hi | y_lo | xx | yy  (~32 MB)
    ushort_t* xhi = (ushort_t*)d_ws;
    ushort_t* xlo = xhi + (size_t)MD;
    ushort_t* yhi = xlo + (size_t)MD;
    ushort_t* ylo = yhi + (size_t)ND;
    float*    xx  = (float*)(ylo + (size_t)ND);
    float*    yy  = xx + M;

    split_bf16_kernel<<<(MD + 255) / 256, 256, 0, stream>>>(x, xhi, xlo, MD);
    split_bf16_kernel<<<(ND + 255) / 256, 256, 0, stream>>>(xt, yhi, ylo, ND);
    rownorm_kernel<<<(M + 7) / 8, 256, 0, stream>>>(x, xx, M);
    rownorm_kernel<<<(N + 7) / 8, 256, 0, stream>>>(xt, yy, N);
    init_out_kernel<<<(M + 255) / 256, 256, 0, stream>>>(out, bias, M);

    const int grid = (M / TILE_M) * NCHUNKS;     // 1024 workgroups
    const size_t shmem = (size_t)(2 * TILE_M * T_STRIDE + 2 * TILE_N * T_STRIDE)
                         * sizeof(ushort_t);     // 266,240 B
    svm_main_kernel<<<grid, 256, shmem, stream>>>(xhi, xlo, yhi, ylo,
                                                  xx, yy, weight, gamma,
                                                  out, M, N);
}
